// Attention_4767413698872
// MI455X (gfx1250) — matrix operand order repaired and checked by reading, not compiled, not run on hardware
//
#include <hip/hip_runtime.h>

typedef __attribute__((ext_vector_type(16))) _Float16 v16h;
typedef __attribute__((ext_vector_type(8)))  _Float16 v8h;
typedef __attribute__((ext_vector_type(8)))  float    v8f;
typedef __attribute__((ext_vector_type(4)))  unsigned int u32x4;
typedef __attribute__((ext_vector_type(8)))  int      i32x8;
typedef __attribute__((ext_vector_type(4)))  int      i32x4;

#define BATCH   4
#define CH      256
#define LEN     4096
#define NGROUPS 8
#define GSIZE   (CH / NGROUPS)   // 32 channels per group
#define EPSV    1e-5f

// ---------------------------------------------------------------------------
// WMMA operand loaders (16x16x32 f16; contiguous k order, shared by A and B, not the table order of CDNA5 ISA 7.12.2)
// A (16x32, MxK) row-major source: lane holds row lane%16; 16 contiguous K
// halves starting at 16*(lane/16).
// B (32x16, KxN) from B^T row-major (n-major) source: lane holds column
// n = lane%16, 16 contiguous K halves starting at 16*(lane/16).
// Works for both global and LDS-resident sources (addrspace inferred).
// ---------------------------------------------------------------------------
__device__ __forceinline__ v16h load_a16(const _Float16* base, int lda) {
  const int lane = threadIdx.x & 31;
  const _Float16* p = base + (size_t)(lane & 15) * lda + ((lane >> 4) * 16);
  v8h lo = *(const v8h*)(p);
  v8h hi = *(const v8h*)(p + 8);
  return __builtin_shufflevector(lo, hi, 0,1,2,3,4,5,6,7,8,9,10,11,12,13,14,15);
}

__device__ __forceinline__ v16h load_b16(const _Float16* base, int ldb) {
  const int lane = threadIdx.x & 31;
  return *(const v16h*)(base + (size_t)(lane & 15) * ldb + ((lane >> 4) * 16));
}

// ---------------------------------------------------------------------------
// TDM: issue one tensor_load_to_lds for a 2-D tile (data_size = 8 bytes).
// Descriptor bitfields per CDNA5 ISA:
//   group0: [1:0]=count=1, [63:32]=lds_addr, [120:64]=global_addr,
//           [127:126]=type=2
//   group1: [17:16]=data_size(3=8B), [79:48]=tensor_dim0, [111:80]=tensor_dim1,
//           [127:112]=tile_dim0, [143:128]=tile_dim1, [207:160]=dim0_stride
// Groups 2/3 zero (<=2-D tensor). Completion tracked with TENSORcnt.
// (6-arg clang-23 builtin form: extra int32x8 group before cpol.)
// ---------------------------------------------------------------------------
__device__ __forceinline__ void tdm_load_2d(unsigned int lds_addr, const void* gptr,
                                            unsigned int tensor_d0, unsigned int tensor_d1,
                                            unsigned int tile_d0, unsigned int tile_d1,
                                            unsigned long long stride0) {
  unsigned long long ga = (unsigned long long)gptr;
  u32x4 g0;
  g0[0] = 1u;                                   // count=1, user descriptor
  g0[1] = lds_addr;                             // LDS byte address
  g0[2] = (unsigned int)ga;                     // global_addr[31:0]
  g0[3] = (unsigned int)(ga >> 32) | (2u << 30);// global_addr[56:32] | type=2
  i32x8 g1;
  g1[0] = (int)(3u << 16);                      // workgroup_mask=0, data_size=8B
  g1[1] = (int)((tensor_d0 & 0xFFFFu) << 16);   // tensor_dim0[15:0]
  g1[2] = (int)((tensor_d0 >> 16) | ((tensor_d1 & 0xFFFFu) << 16));
  g1[3] = (int)((tensor_d1 >> 16) | (tile_d0 << 16));
  g1[4] = (int)tile_d1;                         // tile_dim1 (tile_dim2=0)
  g1[5] = (int)(stride0 & 0xFFFFFFFFu);         // tensor_dim0_stride[31:0]
  g1[6] = (int)((stride0 >> 32) & 0xFFFFu);     // stride0[47:32], dim1_stride=0
  g1[7] = 0;
  i32x4 z4 = {0, 0, 0, 0};
  i32x8 z8 = {0, 0, 0, 0, 0, 0, 0, 0};
  __builtin_amdgcn_tensor_load_to_lds(g0, g1, z4, z4, z8, 0);
}

// ---------------------------------------------------------------------------
// Kernel 1: GroupNorm statistics. One block per (b, g); group = 32*4096 floats
// contiguous. Writes (mu, rstd) pairs.
// ---------------------------------------------------------------------------
__global__ void gn_stats_kernel(const float* __restrict__ x, float* __restrict__ stats) {
  __shared__ float ssum[256];
  __shared__ float ssq[256];
  const int bg = blockIdx.x;
  const float* p = x + (size_t)bg * GSIZE * LEN;
  const int n = GSIZE * LEN;
  float s = 0.f, q = 0.f;
  for (int i = threadIdx.x; i < n; i += 256) { float v = p[i]; s += v; q += v * v; }
  ssum[threadIdx.x] = s; ssq[threadIdx.x] = q;
  __syncthreads();
  for (int d = 128; d > 0; d >>= 1) {
    if (threadIdx.x < d) {
      ssum[threadIdx.x] += ssum[threadIdx.x + d];
      ssq[threadIdx.x]  += ssq[threadIdx.x + d];
    }
    __syncthreads();
  }
  if (threadIdx.x == 0) {
    float mu  = ssum[0] / (float)n;
    float var = ssq[0] / (float)n - mu * mu;
    stats[bg * 2]     = mu;
    stats[bg * 2 + 1] = rsqrtf(var + EPSV);
  }
}

// ---------------------------------------------------------------------------
// Kernel 2: fp32 -> f16 conversion (weights)
// ---------------------------------------------------------------------------
__global__ void cvt_f16_kernel(const float* __restrict__ src, _Float16* __restrict__ dst, int n) {
  int i = blockIdx.x * blockDim.x + threadIdx.x;
  if (i < n) dst[i] = (_Float16)src[i];
}

// ---------------------------------------------------------------------------
// Kernel 3: apply GroupNorm affine and write xn^T [B, L, C] in f16
// ---------------------------------------------------------------------------
__global__ void norm_tr_kernel(const float* __restrict__ x, const float* __restrict__ stats,
                               const float* __restrict__ gw, const float* __restrict__ gb,
                               _Float16* __restrict__ xnt) {
  __shared__ float tile[16][17];
  const int tx = threadIdx.x & 15, ty = threadIdx.x >> 4;
  const int l0 = blockIdx.x * 16, c0 = blockIdx.y * 16, b = blockIdx.z;
  const int c = c0 + ty;
  const int g = c >> 5;  // c / GSIZE
  const float mu = stats[(b * NGROUPS + g) * 2];
  const float rs = stats[(b * NGROUPS + g) * 2 + 1];
  const float v  = x[((size_t)b * CH + c) * LEN + l0 + tx];
  tile[ty][tx] = (v - mu) * rs * gw[c] + gb[c];
  __syncthreads();
  xnt[((size_t)b * LEN + l0 + ty) * CH + c0 + tx] = (_Float16)tile[tx][ty];
}

// ---------------------------------------------------------------------------
// Kernel 4: QKV projection GEMM (one 16x16 tile per wave, K=256 in 8 WMMAs).
// q,k stored transposed-packed as [B,L,C] f16 (q pre-scaled by 1/sqrt(C));
// v stored as [B,C,L] f16 (A-side of the PV GEMM).
// ---------------------------------------------------------------------------
__global__ void qkv_gemm_kernel(const _Float16* __restrict__ wq16, const _Float16* __restrict__ xnt,
                                const float* __restrict__ bqkv,
                                _Float16* __restrict__ qb, _Float16* __restrict__ kb,
                                _Float16* __restrict__ vt) {
  const int lane = threadIdx.x & 31;
  const int tile = blockIdx.x * (blockDim.x >> 5) + (threadIdx.x >> 5);
  const int lt = tile & 255;           // LEN/16 tiles
  const int ot = (tile >> 8) % 48;     // 3C/16 tiles
  const int b  = tile / (256 * 48);
  const int l0 = lt * 16, o0 = ot * 16;
  const _Float16* bsrc = xnt + (size_t)b * LEN * CH + (size_t)l0 * CH;

  v8f acc = {};
#pragma unroll
  for (int s = 0; s < 8; ++s) {
    v16h a  = load_a16(wq16 + (size_t)o0 * CH + 32 * s, CH);
    v16h bb = load_b16(bsrc + 32 * s, CH);
    acc = __builtin_amdgcn_wmma_f32_16x16x32_f16(false, a, false, bb, (short)0, acc, false, false);
  }
  const int n = lane & 15;
  const int mbase = o0 + ((lane >> 4) * 8);
#pragma unroll
  for (int r = 0; r < 8; ++r) acc[r] += bqkv[mbase + r];

  if (o0 < CH) {            // Q: scale by C^-0.5 = 1/16, store [L,C]
    v8h h;
#pragma unroll
    for (int r = 0; r < 8; ++r) h[r] = (_Float16)(acc[r] * 0.0625f);
    *(v8h*)(qb + ((size_t)b * LEN + l0 + n) * CH + mbase) = h;
  } else if (o0 < 2 * CH) { // K: store [L,C]
    v8h h;
#pragma unroll
    for (int r = 0; r < 8; ++r) h[r] = (_Float16)acc[r];
    *(v8h*)(kb + ((size_t)b * LEN + l0 + n) * CH + (mbase - CH)) = h;
  } else {                  // V: store [C,L]
#pragma unroll
    for (int r = 0; r < 8; ++r)
      vt[((size_t)b * CH + (mbase - 2 * CH) + r) * LEN + l0 + n] = (_Float16)acc[r];
  }
}

// ---------------------------------------------------------------------------
// Kernel 5: flash attention with TDM-staged K/V tiles.
// 8 waves per block share batch b and process 8 consecutive 16-row i-tiles.
// Per 32-j slab, wave 0 DMAs K[32,256] (16KB, 1-D) and V^T[256,32] (16KB,
// 256 rows x 64B, row stride 8KB) into double-buffered LDS via
// tensor_load_to_lds; waves consume via ds_load_b128. Softmax runs on S^T
// tiles so the j-reduction is per-lane + one shfl_xor(16).
// ---------------------------------------------------------------------------
__global__ void attn_kernel(const _Float16* __restrict__ qb, const _Float16* __restrict__ kb,
                            const _Float16* __restrict__ vt, _Float16* __restrict__ ao) {
  __shared__ __align__(32) _Float16 kts[2][32 * 256];   // 2 x 16KB
  __shared__ __align__(32) _Float16 vts[2][256 * 32];   // 2 x 16KB

  const int lane = threadIdx.x & 31;
  const int wid  = threadIdx.x >> 5;
  const int wtile = blockIdx.x * (blockDim.x >> 5) + wid;
  const int it = wtile & 255;   // LEN/16
  const int b  = wtile >> 8;    // all 8 waves of a block share b (32 blocks/batch)
  const int i0 = it * 16;
  const _Float16* qp = qb + (size_t)b * LEN * CH;
  const _Float16* kp = kb + (size_t)b * LEN * CH;
  const _Float16* vp = vt + (size_t)b * CH * LEN;

  const unsigned int kts_lds0 = (unsigned int)(size_t)(void*)&kts[0][0];
  const unsigned int kts_lds1 = (unsigned int)(size_t)(void*)&kts[1][0];
  const unsigned int vts_lds0 = (unsigned int)(size_t)(void*)&vts[0][0];
  const unsigned int vts_lds1 = (unsigned int)(size_t)(void*)&vts[1][0];

  // Preload Q as the B operand for all 8 K-steps (64 VGPRs)
  v16h qreg[8];
#pragma unroll
  for (int s = 0; s < 8; ++s) qreg[s] = load_b16(qp + (size_t)i0 * CH + 32 * s, CH);

  v8f oacc[16];
#pragma unroll
  for (int t = 0; t < 16; ++t) { v8f z = {}; oacc[t] = z; }
  float mi = -3.0e38f, li = 0.f;
  const bool hiHalf = lane >= 16;

  // Prologue: stage slab 0 into buffer 0
  if (wid == 0) {
    tdm_load_2d(kts_lds0, kp, 2048u, 1u, 2048u, 0u, 2048ull);            // 16KB 1-D
    tdm_load_2d(vts_lds0, vp, 8u, 256u, 8u, 256u, (unsigned long long)(LEN / 4)); // 256 x 64B
  }

  const int NITER = LEN / 32;
  for (int i = 0; i < NITER; ++i) {
    const int cur = i & 1;
    if (wid == 0) {
      // stage next slab into the other buffer (dummy re-load of slab 0 on last)
      const int jn = (i + 1 < NITER) ? (i + 1) * 32 : 0;
      const unsigned int kd = cur ? kts_lds0 : kts_lds1;
      const unsigned int vd = cur ? vts_lds0 : vts_lds1;
      tdm_load_2d(kd, kp + (size_t)jn * CH, 2048u, 1u, 2048u, 0u, 2048ull);
      tdm_load_2d(vd, vp + jn, 8u, 256u, 8u, 256u, (unsigned long long)(LEN / 4));
      __builtin_amdgcn_s_wait_tensorcnt(2);  // current slab's pair is complete
    }
    __syncthreads();  // current buffers valid for all waves

    const _Float16* kbase = &kts[cur][0];
    const _Float16* vbase = &vts[cur][0];

    // S^T tiles: st1 = slab rows 0..15, st2 = rows 16..31 (cols = i)
    v8f st1 = {}, st2 = {};
#pragma unroll
    for (int s = 0; s < 8; ++s) {
      v16h a1 = load_a16(kbase + 32 * s, CH);
      st1 = __builtin_amdgcn_wmma_f32_16x16x32_f16(false, a1, false, qreg[s], (short)0, st1, false, false);
      v16h a2 = load_a16(kbase + 16 * CH + 32 * s, CH);
      st2 = __builtin_amdgcn_wmma_f32_16x16x32_f16(false, a2, false, qreg[s], (short)0, st2, false, false);
    }
    // column (over j) max for this 32-row slab
    float cm = st1[0];
#pragma unroll
    for (int r = 0; r < 8; ++r) { cm = fmaxf(cm, st1[r]); cm = fmaxf(cm, st2[r]); }
    cm = fmaxf(cm, __shfl_xor(cm, 16, 32));
    const float mn = fmaxf(mi, cm);

    float p1[8], p2[8], cs = 0.f;
#pragma unroll
    for (int r = 0; r < 8; ++r) {
      p1[r] = __expf(st1[r] - mn);
      p2[r] = __expf(st2[r] - mn);
      cs += p1[r] + p2[r];
    }
    cs += __shfl_xor(cs, 16, 32);
    const float alpha = __expf(mi - mn);
    li = li * alpha + cs;
    mi = mn;
#pragma unroll
    for (int t = 0; t < 16; ++t) oacc[t] *= alpha;

    // Build P^T as a B operand (32 j x 16 i): half-wave exchange + f16 pack
    v16h pb = {};
#pragma unroll
    for (int r = 0; r < 8; ++r) {
      float o1 = __shfl_xor(p1[r], 16, 32);
      float o2 = __shfl_xor(p2[r], 16, 32);
      pb[r]     = (_Float16)(hiHalf ? o2 : p1[r]);
      pb[r + 8] = (_Float16)(hiHalf ? p2[r] : o1);
    }
    // O^T[c, i] += V^T[c, j-slab] * P^T
#pragma unroll
    for (int t = 0; t < 16; ++t) {
      v16h av = load_a16(vbase + (16 * t) * 32, 32);
      oacc[t] = __builtin_amdgcn_wmma_f32_16x16x32_f16(false, av, false, pb, (short)0, oacc[t], false, false);
    }
    __syncthreads();  // all waves done with buffer `cur` before it is re-staged
  }

  // normalize and store transposed-packed as [L, C] f16
  const float inv = 1.f / li;
  const int n = lane & 15;
  const int m8 = (lane >> 4) * 8;
  _Float16* dst = ao + ((size_t)b * LEN + i0 + n) * CH;
#pragma unroll
  for (int t = 0; t < 16; ++t) {
    v8h h;
#pragma unroll
    for (int r = 0; r < 8; ++r) h[r] = (_Float16)(oacc[t][r] * inv);
    *(v8h*)(dst + 16 * t + m8) = h;
  }
}

// ---------------------------------------------------------------------------
// Kernel 6: output projection + bias + residual.
// ---------------------------------------------------------------------------
__global__ void out_gemm_kernel(const _Float16* __restrict__ wo16, const _Float16* __restrict__ ao,
                                const float* __restrict__ bout, const float* __restrict__ x,
                                float* __restrict__ out) {
  const int lane = threadIdx.x & 31;
  const int tile = blockIdx.x * (blockDim.x >> 5) + (threadIdx.x >> 5);
  const int lt = tile & 255;
  const int ot = (tile >> 8) & 15;
  const int b  = tile >> 12;
  const int l0 = lt * 16, o0 = ot * 16;
  const _Float16* bsrc = ao + (size_t)b * LEN * CH + (size_t)l0 * CH;

  v8f acc = {};
#pragma unroll
  for (int s = 0; s < 8; ++s) {
    v16h a  = load_a16(wo16 + (size_t)o0 * CH + 32 * s, CH);
    v16h bb = load_b16(bsrc + 32 * s, CH);
    acc = __builtin_amdgcn_wmma_f32_16x16x32_f16(false, a, false, bb, (short)0, acc, false, false);
  }
  const int n = lane & 15;
  const int mbase = o0 + (lane >> 4) * 8;
#pragma unroll
  for (int r = 0; r < 8; ++r) {
    size_t idx = ((size_t)b * CH + mbase + r) * LEN + l0 + n;
    out[idx] = acc[r] + bout[mbase + r] + x[idx];
  }
}

// ---------------------------------------------------------------------------
extern "C" void kernel_launch(void* const* d_in, const int* in_sizes, int n_in,
                              void* d_out, int out_size, void* d_ws, size_t ws_size,
                              hipStream_t stream) {
  (void)in_sizes; (void)n_in; (void)out_size; (void)ws_size;
  const float* x    = (const float*)d_in[0];
  const float* gw   = (const float*)d_in[1];
  const float* gb   = (const float*)d_in[2];
  const float* wqkv = (const float*)d_in[3];
  const float* bqkv = (const float*)d_in[4];
  const float* wout = (const float*)d_in[5];
  const float* bout = (const float*)d_in[6];
  float* out = (float*)d_out;

  // workspace layout (all offsets 512B-aligned)
  char* ws = (char*)d_ws;
  constexpr size_t SZ_WQ  = (size_t)3 * CH * CH * sizeof(_Float16);        // 393216
  constexpr size_t SZ_WO  = (size_t)CH * CH * sizeof(_Float16);           // 131072
  constexpr size_t SZ_MAT = (size_t)BATCH * LEN * CH * sizeof(_Float16);  // 8 MB
  constexpr size_t OFF_WQ  = 1024;
  constexpr size_t OFF_WO  = OFF_WQ + SZ_WQ;
  constexpr size_t OFF_XNT = OFF_WO + SZ_WO;
  constexpr size_t OFF_Q   = OFF_XNT + SZ_MAT;
  constexpr size_t OFF_K   = OFF_Q + SZ_MAT;
  constexpr size_t OFF_VT  = OFF_K + SZ_MAT;
  constexpr size_t OFF_AO  = OFF_VT + SZ_MAT;

  float*    stats = (float*)ws;
  _Float16* wq16  = (_Float16*)(ws + OFF_WQ);
  _Float16* wo16  = (_Float16*)(ws + OFF_WO);
  _Float16* xnt   = (_Float16*)(ws + OFF_XNT);
  _Float16* qbuf  = (_Float16*)(ws + OFF_Q);
  _Float16* kbuf  = (_Float16*)(ws + OFF_K);
  _Float16* vtbuf = (_Float16*)(ws + OFF_VT);
  _Float16* aobuf = (_Float16*)(ws + OFF_AO);

  gn_stats_kernel<<<BATCH * NGROUPS, 256, 0, stream>>>(x, stats);
  cvt_f16_kernel<<<(3 * CH * CH + 255) / 256, 256, 0, stream>>>(wqkv, wq16, 3 * CH * CH);
  cvt_f16_kernel<<<(CH * CH + 255) / 256, 256, 0, stream>>>(wout, wo16, CH * CH);
  norm_tr_kernel<<<dim3(LEN / 16, CH / 16, BATCH), 256, 0, stream>>>(x, stats, gw, gb, xnt);

  // 8 waves per 256-thread block, one 16x16 output tile per wave
  qkv_gemm_kernel<<<(BATCH * (3 * CH / 16) * (LEN / 16)) / 8, 256, 0, stream>>>(
      wq16, xnt, bqkv, qbuf, kbuf, vtbuf);
  attn_kernel<<<(BATCH * (LEN / 16)) / 8, 256, 0, stream>>>(qbuf, kbuf, vtbuf, aobuf);
  out_gemm_kernel<<<(BATCH * (CH / 16) * (LEN / 16)) / 8, 256, 0, stream>>>(
      wo16, aobuf, bout, x, out);
}
